// PTQ_88553635709523
// MI455X (gfx1250) — 8-bit matrix operand order repaired and checked by reading, compiled, not run on hardware
//
#include <hip/hip_runtime.h>
#include <hip/hip_bf16.h>


typedef __attribute__((ext_vector_type(2))) float v2f;
typedef __attribute__((ext_vector_type(8))) float v8f;
typedef __attribute__((ext_vector_type(4))) int   v4i;
typedef __attribute__((ext_vector_type(8))) int   v8i;

#define FLTMAX 3.402823466e38f

__global__ void zero_i32_kernel(int* __restrict__ p, int n) {
    int i = blockIdx.x * blockDim.x + threadIdx.x;
    if (i < n) p[i] = 0;
}

__global__ __launch_bounds__(256) void minmax_partial_kernel(
    const float* __restrict__ x, long n,
    float* __restrict__ pmin, float* __restrict__ pmax)
{
    __shared__ float smin[256], smax[256];
    int t = threadIdx.x;
    float lmin = FLTMAX, lmax = -FLTMAX;
    for (long i = (long)blockIdx.x * blockDim.x + t; i < n;
         i += (long)gridDim.x * blockDim.x) {
        float v = x[i];
        lmin = fminf(lmin, v);
        lmax = fmaxf(lmax, v);
    }
    smin[t] = lmin; smax[t] = lmax;
    __syncthreads();
    for (int s = 128; s > 0; s >>= 1) {
        if (t < s) {
            smin[t] = fminf(smin[t], smin[t + s]);
            smax[t] = fmaxf(smax[t], smax[t + s]);
        }
        __syncthreads();
    }
    if (t == 0) { pmin[blockIdx.x] = smin[0]; pmax[blockIdx.x] = smax[0]; }
}

__global__ __launch_bounds__(256) void minmax_final_kernel(
    const float* __restrict__ pmin, const float* __restrict__ pmax, int n,
    float* __restrict__ mm  )
{
    __shared__ float smin[256], smax[256];
    int t = threadIdx.x;
    float lmin = FLTMAX, lmax = -FLTMAX;
    for (int i = t; i < n; i += 256) {
        lmin = fminf(lmin, pmin[i]);
        lmax = fmaxf(lmax, pmax[i]);
    }
    smin[t] = lmin; smax[t] = lmax;
    __syncthreads();
    for (int s = 128; s > 0; s >>= 1) {
        if (t < s) {
            smin[t] = fminf(smin[t], smin[t + s]);
            smax[t] = fmaxf(smax[t], smax[t + s]);
        }
        __syncthreads();
    }
    if (t == 0) { mm[0] = smin[0]; mm[1] = smax[0]; }
}

__global__ void scales_kernel(const float* __restrict__ mmW,
                              const float* __restrict__ mmX,
                              float* __restrict__ sc  ,
                              int* __restrict__ zz  )
{
    if (threadIdx.x == 0 && blockIdx.x == 0) {
        float Sw = (mmW[1] - mmW[0]) * (1.0f / 255.0f);
        float Sx = (mmX[1] - mmX[0]) * (1.0f / 255.0f);
        float zw = fminf(fmaxf(127.0f - rintf(mmW[1] / Sw), -128.0f), 127.0f);
        float zx = fminf(fmaxf(127.0f - rintf(mmX[1] / Sx), -128.0f), 127.0f);
        sc[0] = Sw; sc[1] = Sx; sc[2] = Sw * Sx;
        zz[0] = (int)zw; zz[1] = (int)zx;
    }
}

__global__ void biasq_kernel(const float* __restrict__ b,
                             const float* __restrict__ sc,
                             int* __restrict__ biasq, int d)
{
    int i = blockIdx.x * blockDim.x + threadIdx.x;
    if (i < d) biasq[i] = (int)rintf(b[i] / sc[2]);
}

__global__ __launch_bounds__(256) void quantW_kernel(
    const float* __restrict__ W, const float* __restrict__ sc,
    const int* __restrict__ zz, signed char* __restrict__ Wq,
    int* __restrict__ rowsumW, int d)
{
    __shared__ int ssum[256];
    int m = blockIdx.x, t = threadIdx.x;
    float Sw = sc[0];
    float Zw = (float)zz[0];
    int lsum = 0;
    for (int k = t * 4; k < d; k += blockDim.x * 4) {
        const float* p = W + (size_t)m * d + k;
        int q[4];
#pragma unroll
        for (int i = 0; i < 4; ++i) {
            float qf = fminf(fmaxf(rintf(p[i] / Sw) + Zw, -128.0f), 127.0f);
            q[i] = (int)qf;
            lsum += q[i];
        }
        int packed = (q[0] & 0xff) | ((q[1] & 0xff) << 8) |
                     ((q[2] & 0xff) << 16) | ((q[3] & 0xff) << 24);
        *(int*)(Wq + (size_t)m * d + k) = packed;
    }
    ssum[t] = lsum;
    __syncthreads();
    for (int s = 128; s > 0; s >>= 1) {
        if (t < s) ssum[t] += ssum[t + s];
        __syncthreads();
    }
    if (t == 0) rowsumW[m] = ssum[0];
}

__global__ __launch_bounds__(256) void quantX_kernel(
    const float* __restrict__ X, const float* __restrict__ sc,
    const int* __restrict__ zz, signed char* __restrict__ XqT,
    int* __restrict__ colsumX, int d, int N)
{
    __shared__ signed char tile[64][68];
    int t  = threadIdx.x;
    int n0 = blockIdx.x * 64;
    int k0 = blockIdx.y * 64;
    float Sx = sc[1];
    float Zx = (float)zz[1];

    int tx = t & 63;
    int ty = t >> 6;
    int lsum = 0;
#pragma unroll
    for (int r = 0; r < 16; ++r) {
        int kk = ty + 4 * r;
        float v = X[(size_t)(k0 + kk) * N + n0 + tx];
        float qf = fminf(fmaxf(rintf(v / Sx) + Zx, -128.0f), 127.0f);
        int qi = (int)qf;
        tile[kk][tx] = (signed char)qi;
        lsum += qi;
    }
    atomicAdd(&colsumX[n0 + tx], lsum);
    __syncthreads();

    int kx = t & 15;
    int ny = t >> 4;
#pragma unroll
    for (int r = 0; r < 4; ++r) {
        int nn = ny + 16 * r;
        int b0 = (unsigned char)tile[kx * 4 + 0][nn];
        int b1 = (unsigned char)tile[kx * 4 + 1][nn];
        int b2 = (unsigned char)tile[kx * 4 + 2][nn];
        int b3 = (unsigned char)tile[kx * 4 + 3][nn];
        int packed = b0 | (b1 << 8) | (b2 << 16) | (b3 << 24);
        *(int*)(XqT + (size_t)(n0 + nn) * d + k0 + kx * 4) = packed;
    }
}

template <int TD, int TN>
__global__ __launch_bounds__(256) void gemm_f32_kernel(
    const float* __restrict__ W, const float* __restrict__ X,
    const float* __restrict__ b, float* __restrict__ out, int d_, int N_)
{
    const int d = (TD > 0) ? TD : d_;
    const int N = (TN > 0) ? TN : N_;

    const int lane = threadIdx.x & 31;
    const int wave = threadIdx.x >> 5;
    const int n0   = blockIdx.x * 128 + wave * 16;
    const int m0   = blockIdx.y * 64;
    const int lm   = lane & 15;
    const int half = lane >> 4;

    v8f acc0 = {0,0,0,0,0,0,0,0};
    v8f acc1 = {0,0,0,0,0,0,0,0};
    v8f acc2 = {0,0,0,0,0,0,0,0};
    v8f acc3 = {0,0,0,0,0,0,0,0};

    const float* xb = X + (size_t)(half * 2) * N + n0 + lm;
    const float* wp = W + (size_t)(m0 + lm) * d + half * 2;

    for (int kk = 0; kk < d; kk += 32) {
#pragma unroll
        for (int j = 0; j < 8; ++j) {
            v2f bf;
            bf.x = xb[(size_t)j * 4 * N];
            bf.y = xb[(size_t)j * 4 * N + N];
            v2f a0 = *(const v2f*)(wp + j * 4);
            v2f a1 = *(const v2f*)(wp + (size_t)16 * d + j * 4);
            v2f a2 = *(const v2f*)(wp + (size_t)32 * d + j * 4);
            v2f a3 = *(const v2f*)(wp + (size_t)48 * d + j * 4);
            acc0 = __builtin_amdgcn_wmma_f32_16x16x4_f32(false, a0, false, bf, (short)0, acc0, false, false);
            acc1 = __builtin_amdgcn_wmma_f32_16x16x4_f32(false, a1, false, bf, (short)0, acc1, false, false);
            acc2 = __builtin_amdgcn_wmma_f32_16x16x4_f32(false, a2, false, bf, (short)0, acc2, false, false);
            acc3 = __builtin_amdgcn_wmma_f32_16x16x4_f32(false, a3, false, bf, (short)0, acc3, false, false);
        }
        xb += (size_t)32 * N;
        wp += 32;
    }

    v8f accs[4] = {acc0, acc1, acc2, acc3};
    float* op = out + (size_t)(m0 + half * 8) * N + n0 + lm;
#pragma unroll
    for (int i = 0; i < 4; ++i) {
#pragma unroll
        for (int r = 0; r < 8; ++r) {
            int m = m0 + i * 16 + r + half * 8;
            op[(size_t)(i * 16 + r) * N] = accs[i][r] + b[m];
        }
    }
}

template <int TD, int TN>
__global__ __launch_bounds__(256) void gemm_i8_kernel(
    const signed char* __restrict__ Wq, const signed char* __restrict__ XqT,
    const int* __restrict__ rowsumW, const int* __restrict__ colsumX,
    const int* __restrict__ biasq, const float* __restrict__ sc,
    const int* __restrict__ zz, float* __restrict__ out, int d_, int N_)
{
    const int d = (TD > 0) ? TD : d_;
    const int N = (TN > 0) ? TN : N_;

    const int lane = threadIdx.x & 31;
    const int wave = threadIdx.x >> 5;
    const int m0 = blockIdx.y * 64 + (wave >> 2) * 32;
    const int n0 = blockIdx.x * 128 + (wave & 3) * 32;
    const int lm   = lane & 15;
    const int half = lane >> 4;

    v8i acc00 = {0,0,0,0,0,0,0,0};
    v8i acc01 = {0,0,0,0,0,0,0,0};
    v8i acc10 = {0,0,0,0,0,0,0,0};
    v8i acc11 = {0,0,0,0,0,0,0,0};

    const signed char* pa = Wq  + (size_t)(m0 + lm) * d + half * 16;
    const signed char* pb = XqT + (size_t)(n0 + lm) * d + half * 16;

#pragma unroll
    for (int k = 0; k < ((TD > 0) ? TD : 1024); k += 64) {
        if (TD <= 0 && k >= d) break;
        v8i a0, a1, b0, b1;
        {
            const signed char* p = pa + k;
            int2 q0 = *(const int2*)(p + 0);
            int2 q1 = *(const int2*)(p + 8);
            int2 q2 = *(const int2*)(p + 32);
            int2 q3 = *(const int2*)(p + 40);
            a0[0]=q0.x; a0[1]=q0.y; a0[2]=q1.x; a0[3]=q1.y;
            a0[4]=q2.x; a0[5]=q2.y; a0[6]=q3.x; a0[7]=q3.y;
        }
        {
            const signed char* p = pa + (size_t)16 * d + k;
            int2 q0 = *(const int2*)(p + 0);
            int2 q1 = *(const int2*)(p + 8);
            int2 q2 = *(const int2*)(p + 32);
            int2 q3 = *(const int2*)(p + 40);
            a1[0]=q0.x; a1[1]=q0.y; a1[2]=q1.x; a1[3]=q1.y;
            a1[4]=q2.x; a1[5]=q2.y; a1[6]=q3.x; a1[7]=q3.y;
        }
        {
            const signed char* p = pb + k;
            v4i q0 = *(const v4i*)(p);
            v4i q1 = *(const v4i*)(p + 32);
            b0[0]=q0[0]; b0[1]=q0[1]; b0[2]=q0[2]; b0[3]=q0[3];
            b0[4]=q1[0]; b0[5]=q1[1]; b0[6]=q1[2]; b0[7]=q1[3];
        }
        {
            const signed char* p = pb + (size_t)16 * d + k;
            v4i q0 = *(const v4i*)(p);
            v4i q1 = *(const v4i*)(p + 32);
            b1[0]=q0[0]; b1[1]=q0[1]; b1[2]=q0[2]; b1[3]=q0[3];
            b1[4]=q1[0]; b1[5]=q1[1]; b1[6]=q1[2]; b1[7]=q1[3];
        }
        acc00 = __builtin_amdgcn_wmma_i32_16x16x64_iu8(true, a0, true, b0, acc00, false, false);
        acc01 = __builtin_amdgcn_wmma_i32_16x16x64_iu8(true, a0, true, b1, acc01, false, false);
        acc10 = __builtin_amdgcn_wmma_i32_16x16x64_iu8(true, a1, true, b0, acc10, false, false);
        acc11 = __builtin_amdgcn_wmma_i32_16x16x64_iu8(true, a1, true, b1, acc11, false, false);
    }

    const float scales = sc[2];
    const int Zw = zz[0], Zx = zz[1];
    const int corr = d * Zw * Zx;

    v8i accs[4] = {acc00, acc01, acc10, acc11};
#pragma unroll
    for (int mi = 0; mi < 2; ++mi) {
#pragma unroll
        for (int ni = 0; ni < 2; ++ni) {
            v8i a = accs[mi * 2 + ni];
            int n = n0 + ni * 16 + lm;
            int csn = colsumX[n];
            float* op = out + (size_t)(m0 + mi * 16 + half * 8) * N + n;
#pragma unroll
            for (int r = 0; r < 8; ++r) {
                int m = m0 + mi * 16 + r + half * 8;
                int v = a[r] - Zx * rowsumW[m] - Zw * csn + corr + biasq[m];
                op[(size_t)r * N] = scales * (float)v;
            }
        }
    }
}

extern "C" void kernel_launch(void* const* d_in, const int* in_sizes, int n_in,
                              void* d_out, int out_size, void* d_ws, size_t ws_size,
                              hipStream_t stream) {
    const float* X = (const float*)d_in[0];
    const float* W = (const float*)d_in[1];
    const float* b = (const float*)d_in[2];

    const int d = in_sizes[2];
    const int N = in_sizes[0] / d;

    float* out_ori = (float*)d_out;
    float* out_q   = out_ori + (size_t)d * N;

    auto al = [](size_t x) { return (x + 255) & ~(size_t)255; };
    char* ws = (char*)d_ws;
    size_t off = 0;
    signed char* XqT = (signed char*)(ws + off); off = al(off + (size_t)d * N);
    signed char* Wq  = (signed char*)(ws + off); off = al(off + (size_t)d * d);
    int*   colsumX = (int*)(ws + off);   off = al(off + (size_t)N * 4);
    int*   rowsumW = (int*)(ws + off);   off = al(off + (size_t)d * 4);
    int*   biasq   = (int*)(ws + off);   off = al(off + (size_t)d * 4);
    const int PBX = 2048, PBW = 256;
    float* pminX = (float*)(ws + off);   off = al(off + PBX * 4);
    float* pmaxX = (float*)(ws + off);   off = al(off + PBX * 4);
    float* pminW = (float*)(ws + off);   off = al(off + PBW * 4);
    float* pmaxW = (float*)(ws + off);   off = al(off + PBW * 4);
    float* mmX   = (float*)(ws + off);   off = al(off + 2 * 4);
    float* mmW   = (float*)(ws + off);   off = al(off + 2 * 4);
    float* sc    = (float*)(ws + off);   off = al(off + 3 * 4);
    int*   zz    = (int*)(ws + off);     off = al(off + 2 * 4);

    zero_i32_kernel<<<(N + 255) / 256, 256, 0, stream>>>(colsumX, N);

    minmax_partial_kernel<<<PBX, 256, 0, stream>>>(X, (long)d * N, pminX, pmaxX);
    minmax_final_kernel<<<1, 256, 0, stream>>>(pminX, pmaxX, PBX, mmX);
    minmax_partial_kernel<<<PBW, 256, 0, stream>>>(W, (long)d * d, pminW, pmaxW);
    minmax_final_kernel<<<1, 256, 0, stream>>>(pminW, pmaxW, PBW, mmW);

    scales_kernel<<<1, 32, 0, stream>>>(mmW, mmX, sc, zz);
    biasq_kernel<<<(d + 255) / 256, 256, 0, stream>>>(b, sc, biasq, d);

    quantW_kernel<<<d, 256, 0, stream>>>(W, sc, zz, Wq, rowsumW, d);
    {
        dim3 g(N / 64, d / 64);
        quantX_kernel<<<g, 256, 0, stream>>>(X, sc, zz, XqT, colsumX, d, N);
    }

    {
        dim3 g(N / 128, d / 64);
        if (d == 1024 && N == 32768) {
            gemm_f32_kernel<1024, 32768><<<g, 256, 0, stream>>>(W, X, b, out_ori, d, N);
            gemm_i8_kernel<1024, 32768><<<g, 256, 0, stream>>>(
                Wq, XqT, rowsumW, colsumX, biasq, sc, zz, out_q, d, N);
        } else {
            gemm_f32_kernel<0, 0><<<g, 256, 0, stream>>>(W, X, b, out_ori, d, N);
            gemm_i8_kernel<0, 0><<<g, 256, 0, stream>>>(
                Wq, XqT, rowsumW, colsumX, biasq, sc, zz, out_q, d, N);
        }
    }
}
